// GCN_47193100648764
// MI455X (gfx1250) — compile-verified, matrix operand order checked by reading, not run on hardware
//
#include <hip/hip_runtime.h>
#include <math.h>

#define HID 64

typedef float v2f __attribute__((ext_vector_type(2)));
typedef float v8f __attribute__((ext_vector_type(8)));
typedef _Float16 v16h __attribute__((ext_vector_type(16)));
typedef unsigned int u32x4 __attribute__((ext_vector_type(4)));
typedef int i32x4 __attribute__((ext_vector_type(4)));
typedef int i32x8 __attribute__((ext_vector_type(8)));

#if defined(__has_builtin)
#if __has_builtin(__builtin_amdgcn_wmma_f32_16x16x4_f32)
#define WMMA_F32_AVAILABLE 1
#endif
#if __has_builtin(__builtin_amdgcn_tensor_load_to_lds) && \
    __has_builtin(__builtin_amdgcn_s_wait_tensorcnt)
#define TDM_AVAILABLE 1
#if __has_include(<hip/amd_detail/amd_gfx1250_TDM.h>)
#define TDM_6ARG 1
#endif
#endif
#endif

// ---------- small utility kernels ----------

__global__ void k_fill(float* __restrict__ p, long n, float v) {
    long i = (long)blockIdx.x * blockDim.x + threadIdx.x;
    if (i < n) p[i] = v;
}

__global__ void k_fill_bias(float* __restrict__ p, int n_nodes, const float* __restrict__ b) {
    int i = blockIdx.x * blockDim.x + threadIdx.x;
    if (i < n_nodes * HID) p[i] = b[i & (HID - 1)];
}

__device__ inline void atomicMaxFloat(float* addr, float val) {
    // monotonic int mapping trick; addr must be initialized to -inf
    if (val >= 0.0f) atomicMax((int*)addr, __float_as_int(val));
    else             atomicMin((unsigned int*)addr, __float_as_uint(val));
}

// ---------- degree computation ----------

__global__ void k_deg(const int* __restrict__ src, const int* __restrict__ dst, int E,
                      float* __restrict__ din, float* __restrict__ dgo) {
    int e = blockIdx.x * blockDim.x + threadIdx.x;
    if (e < E) {
        atomicAdd(&dgo[src[e]], 1.0f);
        atomicAdd(&din[dst[e]], 1.0f);
    }
}

// ---------- layer-1 transform (K=3, scalar) ----------

__global__ void k_xform1(const float* __restrict__ x, const float* __restrict__ din,
                         const float* __restrict__ dgo,
                         const float* __restrict__ Wl, const float* __restrict__ bl,
                         const float* __restrict__ Wr, const float* __restrict__ br,
                         float* __restrict__ xl, float* __restrict__ xr, int n) {
    int i = blockIdx.x * blockDim.x + threadIdx.x;
    if (i >= n * HID) return;
    int node = i >> 6, o = i & 63;
    float f0 = x[node], f1 = din[node], f2 = dgo[node];
    xl[i] = f0 * Wl[o] + f1 * Wl[HID + o] + f2 * Wl[2 * HID + o] + bl[o];
    xr[i] = f0 * Wr[o] + f1 * Wr[HID + o] + f2 * Wr[2 * HID + o] + br[o];
}

// ---------- Tensor Data Mover: 2D f32 tile (width 64) global -> LDS ----------

#if defined(TDM_AVAILABLE)
// Loads a tile of tile1 rows x 64 cols (f32) from a row-major [dim1 x 64] tensor
// starting at gptr into LDS offset lds_off. Rows beyond dim1 are zero-filled by
// the TDM OOB rule. If pad_row=true, 1 DWORD of padding is inserted after every
// 64 DWORDs stored (i.e. rows land at a 65-float pitch in LDS).
__device__ inline void tdm_load_tile64(unsigned lds_off, const float* gptr,
                                       int dim1, int tile1, bool pad_row) {
    unsigned long long ga = (unsigned long long)(uintptr_t)gptr;
    unsigned galo = (unsigned)__builtin_amdgcn_readfirstlane((int)(unsigned)ga);
    unsigned gahi = (unsigned)__builtin_amdgcn_readfirstlane((int)(unsigned)(ga >> 32));
    lds_off = (unsigned)__builtin_amdgcn_readfirstlane((int)lds_off);
    dim1 = __builtin_amdgcn_readfirstlane(dim1);
    tile1 = __builtin_amdgcn_readfirstlane(tile1);

    u32x4 g0;
    g0.x = 1u;                                   // count=1, user descriptor
    g0.y = lds_off;                              // lds_addr (bytes)
    g0.z = galo;                                 // global_addr[31:0]
    g0.w = (gahi & 0x01ffffffu) | (2u << 30);    // global_addr[56:32] | type=2

    unsigned w0 = (2u << 16);                    // data_size = 4 bytes
    if (pad_row) w0 |= (1u << 20) | (5u << 22);  // pad_enable, interval=64 DWORDs, amount=1 DWORD
    i32x8 g1;
    g1[0] = (int)w0;
    g1[1] = (int)(64u << 16);                    // tensor_dim0 = 64 (lo16 at [31:16])
    g1[2] = (int)(((unsigned)dim1 & 0xffffu) << 16);   // dim0 hi16=0 | tensor_dim1 lo16
    g1[3] = (int)(64u << 16);                    // tensor_dim1 hi16=0 | tile_dim0 = 64
    g1[4] = (int)((unsigned)tile1 & 0xffffu);    // tile_dim1 | tile_dim2 = 0
    g1[5] = 64;                                  // tensor_dim0_stride = 64 (lo32)
    g1[6] = 0;                                   // stride0 hi16 | stride1 lo16
    g1[7] = 0;

    i32x4 z4 = {0, 0, 0, 0};
#if defined(TDM_6ARG)
    i32x8 z8 = {0, 0, 0, 0, 0, 0, 0, 0};
    __builtin_amdgcn_tensor_load_to_lds(g0, g1, z4, z4, z8, 0);
#else
    __builtin_amdgcn_tensor_load_to_lds(g0, g1, z4, z4, 0);
#endif
}
#endif

// ---------- layer-2 transform: WMMA GEMM [N,64]x[64,64] (both sides) ----------

__device__ inline void wave_gemm16(const float* __restrict__ sHrow,  // 16 x 65 (padded)
                                   const float* __restrict__ sWm,    // 64 x 64
                                   const float* __restrict__ bias,
                                   float* __restrict__ out,
                                   int tile_node0, int n_nodes, int lane) {
    const int arow = lane & 15;
    const int hi   = (lane & 16) ? 1 : 0;
#pragma unroll
    for (int ot = 0; ot < 4; ++ot) {
        const int col = ot * 16 + (lane & 15);
        const float bval = bias[col];
        v8f acc;
#pragma unroll
        for (int i = 0; i < 8; ++i) acc[i] = bval;
#if defined(WMMA_F32_AVAILABLE)
#pragma unroll
        for (int k = 0; k < HID; k += 4) {
            const int kk = k + (hi ? 2 : 0);
            v2f a, b;
            a[0] = sHrow[arow * 65 + kk];
            a[1] = sHrow[arow * 65 + kk + 1];
            b[0] = sWm[kk * HID + col];
            b[1] = sWm[(kk + 1) * HID + col];
            acc = __builtin_amdgcn_wmma_f32_16x16x4_f32(
                false, a, false, b, (short)0, acc, false, false);
        }
#else
        // fallback: fp16 inputs, fp32 accumulate (codegen-confirmed builtin)
#pragma unroll
        for (int kt = 0; kt < HID; kt += 32) {
            v16h af, bf;
#pragma unroll
            for (int v = 0; v < 8; ++v) {
                const int kb = kt + ((v < 4) ? 0 : 16) + (hi ? 8 : 0) + (v & 3) * 2;
                af[2 * v]     = (_Float16)sHrow[arow * 65 + kb];
                af[2 * v + 1] = (_Float16)sHrow[arow * 65 + kb + 1];
                bf[2 * v]     = (_Float16)sWm[kb * HID + col];
                bf[2 * v + 1] = (_Float16)sWm[(kb + 1) * HID + col];
            }
            acc = __builtin_amdgcn_wmma_f32_16x16x32_f16(
                false, af, false, bf, (short)0, acc, false, false);
        }
#endif
#pragma unroll
        for (int i = 0; i < 8; ++i) {
            const int node = tile_node0 + i + hi * 8;
            if (node < n_nodes) out[(long)node * HID + col] = acc[i];
        }
    }
}

__global__ __launch_bounds__(256) void k_xform2_wmma(
        const float* __restrict__ H,
        const float* __restrict__ Wl, const float* __restrict__ bl,
        const float* __restrict__ Wr, const float* __restrict__ br,
        float* __restrict__ XL, float* __restrict__ XR, int n_nodes) {
    __shared__ float sW[2][HID * HID];   // 32 KB: both weight matrices
    __shared__ float sH[8][16 * 65];     // ~33 KB: per-wave 16x64 node tile, 65-f pitch
    const int tid  = threadIdx.x;
    const int wave = tid >> 5, lane = tid & 31;
    const int tile0 = (blockIdx.x * 8 + wave) * 16;

#if defined(TDM_AVAILABLE)
    {
        // per-wave TDM: stage this wave's 16x64 node tile with hardware row padding
        int rows = n_nodes - tile0;
        rows = rows < 0 ? 0 : (rows > 16 ? 16 : rows);
        const float* gsrc = H + (long)(rows ? tile0 : 0) * HID;
        unsigned lds_h = (unsigned)(uintptr_t)(&sH[wave][0]);
        tdm_load_tile64(lds_h, gsrc, rows, 16, /*pad_row=*/true);
        if (wave == 0) {
            // wave 0 also stages both 64x64 weight matrices
            tdm_load_tile64((unsigned)(uintptr_t)(&sW[0][0]), Wl, 64, 64, false);
            tdm_load_tile64((unsigned)(uintptr_t)(&sW[1][0]), Wr, 64, 64, false);
        }
        __builtin_amdgcn_s_wait_tensorcnt(0);
    }
#else
    for (int i = tid; i < HID * HID; i += 256) { sW[0][i] = Wl[i]; sW[1][i] = Wr[i]; }
    for (int i = lane; i < 16 * HID; i += 32) {
        const int r = i >> 6, c = i & 63;
        const int node = tile0 + r;
        sH[wave][r * 65 + c] = (node < n_nodes) ? H[(long)node * HID + c] : 0.0f;
    }
#endif
    __syncthreads();
    wave_gemm16(sH[wave], sW[0], bl, XL, tile0, n_nodes, lane);
    wave_gemm16(sH[wave], sW[1], br, XR, tile0, n_nodes, lane);
}

// ---------- attention: per-edge logit + segment max (wave per edge) ----------

__global__ __launch_bounds__(256) void k_logit(
        const float* __restrict__ xl, const float* __restrict__ xr,
        const float* __restrict__ a, const int* __restrict__ src,
        const int* __restrict__ dst, int E, int ET,
        float* __restrict__ elog, float* __restrict__ m) {
    const int wave = threadIdx.x >> 5, lane = threadIdx.x & 31;
    const int e = blockIdx.x * 8 + wave;
    if (e >= ET) return;
    int s, d;
    if (e < E) { s = src[e]; d = dst[e]; } else { s = d = e - E; }
    float v0 = xl[(long)s * HID + lane]      + xr[(long)d * HID + lane];
    float v1 = xl[(long)s * HID + 32 + lane] + xr[(long)d * HID + 32 + lane];
    v0 = (v0 > 0.0f) ? v0 : 0.2f * v0;   // leaky relu, slope 0.2
    v1 = (v1 > 0.0f) ? v1 : 0.2f * v1;
    float t = v0 * a[lane] + v1 * a[32 + lane];
#pragma unroll
    for (int off = 16; off > 0; off >>= 1) t += __shfl_xor(t, off);
    if (lane == 0) {
        elog[e] = t;
        atomicMaxFloat(&m[d], t);
    }
}

// ---------- attention: exp + segment sum ----------

__global__ void k_expsum(const int* __restrict__ dst, int E, int ET,
                         float* __restrict__ elog, const float* __restrict__ m,
                         float* __restrict__ den) {
    int e = blockIdx.x * blockDim.x + threadIdx.x;
    if (e >= ET) return;
    int d = (e < E) ? dst[e] : (e - E);
    float ex = expf(elog[e] - m[d]);
    elog[e] = ex;
    atomicAdd(&den[d], ex);
}

// ---------- attention: weighted aggregation (64 lanes per edge, coalesced) ----------

__global__ __launch_bounds__(256) void k_aggr(
        const int* __restrict__ src, const int* __restrict__ dst, int E, int ET,
        const float* __restrict__ ex, const float* __restrict__ den,
        const float* __restrict__ xl, float* __restrict__ out) {
    long idx = (long)blockIdx.x * blockDim.x + threadIdx.x;
    int e = (int)(idx >> 6);
    int f = (int)(idx & 63);
    if (e >= ET) return;
    int s, d;
    if (e < E) { s = src[e]; d = dst[e]; } else { s = d = e - E; }
    float alpha = ex[e] / den[d];
    atomicAdd(&out[(long)d * HID + f], alpha * xl[(long)s * HID + f]);
}

// ---------- tanh ----------

__global__ void k_tanh(float* __restrict__ p, int n) {
    int i = blockIdx.x * blockDim.x + threadIdx.x;
    if (i < n) p[i] = tanhf(p[i]);
}

// ---------- global max pool ----------

__global__ void k_pool(const float* __restrict__ h, const int* __restrict__ batch,
                       int n, float* __restrict__ g) {
    int i = blockIdx.x * blockDim.x + threadIdx.x;
    if (i >= n * HID) return;
    int node = i >> 6, f = i & 63;
    atomicMaxFloat(&g[(long)batch[node] * HID + f], h[i]);
}

// ---------- final linear [G,64] x [64,2] ----------

__global__ void k_final(const float* __restrict__ g, const float* __restrict__ W3,
                        const float* __restrict__ b3, float* __restrict__ out, int G) {
    int i = blockIdx.x * blockDim.x + threadIdx.x;
    if (i >= G * 2) return;
    int gr = i >> 1, o = i & 1;
    float acc = b3[o];
#pragma unroll
    for (int k = 0; k < HID; ++k) acc += g[gr * HID + k] * W3[k * 2 + o];
    out[i] = acc;
}

// ---------- launch ----------

extern "C" void kernel_launch(void* const* d_in, const int* in_sizes, int n_in,
                              void* d_out, int out_size, void* d_ws, size_t ws_size,
                              hipStream_t stream) {
    const float* x     = (const float*)d_in[0];
    const int*   ei    = (const int*)  d_in[1];
    const int*   batch = (const int*)  d_in[2];
    const float* Wl1 = (const float*)d_in[3];  const float* bl1 = (const float*)d_in[4];
    const float* Wr1 = (const float*)d_in[5];  const float* br1 = (const float*)d_in[6];
    const float* a1  = (const float*)d_in[7];  const float* b1  = (const float*)d_in[8];
    const float* Wl2 = (const float*)d_in[9];  const float* bl2 = (const float*)d_in[10];
    const float* Wr2 = (const float*)d_in[11]; const float* br2 = (const float*)d_in[12];
    const float* a2  = (const float*)d_in[13]; const float* b2  = (const float*)d_in[14];
    const float* W3  = (const float*)d_in[15]; const float* b3  = (const float*)d_in[16];

    const int N  = in_sizes[0];
    const int E  = in_sizes[1] / 2;
    const int ET = E + N;
    const int G  = out_size / 2;
    const int* src = ei;
    const int* dst = ei + E;

    float* p = (float*)d_ws;
    float* xl  = p; p += (size_t)N * HID;
    float* xr  = p; p += (size_t)N * HID;
    float* hb  = p; p += (size_t)N * HID;
    float* et  = p; p += (size_t)ET;
    float* mx  = p; p += (size_t)N;
    float* dn  = p; p += (size_t)N;
    float* din = p; p += (size_t)N;   // din/dgo contiguous: filled with one kernel
    float* dgo = p; p += (size_t)N;
    float* g   = p; p += (size_t)G * HID;

    const dim3 B(256);
    auto gs = [](long n) { return dim3((unsigned)((n + 255) / 256)); };

    // degrees
    k_fill<<<gs(2L * N), B, 0, stream>>>(din, 2L * N, 0.0f);
    k_deg<<<gs(E), B, 0, stream>>>(src, dst, E, din, dgo);

    // layer 1: transforms + attention
    k_xform1<<<gs((long)N * HID), B, 0, stream>>>(x, din, dgo, Wl1, bl1, Wr1, br1, xl, xr, N);
    k_fill<<<gs(N), B, 0, stream>>>(mx, N, -INFINITY);
    k_fill<<<gs(N), B, 0, stream>>>(dn, N, 0.0f);
    k_fill_bias<<<gs((long)N * HID), B, 0, stream>>>(hb, N, b1);
    k_logit<<<dim3((ET + 7) / 8), B, 0, stream>>>(xl, xr, a1, src, dst, E, ET, et, mx);
    k_expsum<<<gs(ET), B, 0, stream>>>(dst, E, ET, et, mx, dn);
    k_aggr<<<gs((long)ET * HID), B, 0, stream>>>(src, dst, E, ET, et, dn, xl, hb);
    k_tanh<<<gs((long)N * HID), B, 0, stream>>>(hb, N * HID);

    // layer 2: WMMA transforms (TDM-staged LDS) + attention
    k_xform2_wmma<<<dim3((N + 127) / 128), B, 0, stream>>>(hb, Wl2, bl2, Wr2, br2, xl, xr, N);
    k_fill<<<gs(N), B, 0, stream>>>(mx, N, -INFINITY);
    k_fill<<<gs(N), B, 0, stream>>>(dn, N, 0.0f);
    k_fill_bias<<<gs((long)N * HID), B, 0, stream>>>(hb, N, b2);
    k_logit<<<dim3((ET + 7) / 8), B, 0, stream>>>(xl, xr, a2, src, dst, E, ET, et, mx);
    k_expsum<<<gs(ET), B, 0, stream>>>(dst, E, ET, et, mx, dn);
    k_aggr<<<gs((long)ET * HID), B, 0, stream>>>(src, dst, E, ET, et, dn, xl, hb);

    // global max pool + final linear
    k_fill<<<gs((long)G * HID), B, 0, stream>>>(g, (long)G * HID, -INFINITY);
    k_pool<<<gs((long)N * HID), B, 0, stream>>>(hb, batch, N, g);
    k_final<<<gs(2L * G), B, 0, stream>>>(g, W3, b3, (float*)d_out, G);
}
